// MLA_68324339745446
// MI455X (gfx1250) — matrix operand order repaired and checked by reading, not compiled, not run on hardware
//
#include <hip/hip_runtime.h>
#include <hip/hip_bf16.h>
#include <stdint.h>

typedef __attribute__((ext_vector_type(16))) _Float16 v16h;
typedef __attribute__((ext_vector_type(8)))  _Float16 v8h;
typedef __attribute__((ext_vector_type(8)))  float    v8f;

#define B_   2
#define T_   2048
#define DIM_ 2048
#define H_   16
#define QR_  768
#define KVR_ 512
#define DN_  128
#define DR_  64
#define DV_  128
#define BT_  (B_*T_)
#define HD_  192                 /* DN + DR */

// ---------------- fragment loaders (16-bit A/B layouts, ISA 7.12.2) ----------------
// A 16x32: lanes 0-15 hold M=lane, K {0..7,16..23}; lanes 16-31 hold K {8..15,24..31}
__device__ inline v16h frag_a(const _Float16* p0, int ld) {
  int lane = threadIdx.x & 31;
  const _Float16* p = p0 + (size_t)(lane & 15) * ld + ((lane >> 4) << 3);
  v8h lo = *(const v8h*)p;
  v8h hi = *(const v8h*)(p + 16);
  v16h r;
#pragma unroll
  for (int i = 0; i < 8; ++i) { r[i] = lo[i]; r[i + 8] = hi[i]; }
  return r;
}
// B 32x16 (stored as B^T row-major, i.e. N rows x K cols): lane n holds col n,
// lanes 0-15 K {0..7,16..23}; lanes 16-31 K {8..15,24..31}
__device__ inline v16h frag_b(const _Float16* p0, int ld) {
  int lane = threadIdx.x & 31;
  const _Float16* p = p0 + (size_t)(lane & 15) * ld + ((lane >> 4) << 3);
  v8h lo = *(const v8h*)p;
  v8h hi = *(const v8h*)(p + 16);
  v16h r;
#pragma unroll
  for (int i = 0; i < 8; ++i) { r[i] = lo[i]; r[i + 8] = hi[i]; }
  return r;
}

// async 16B global->LDS copy (per lane), tracked by ASYNCcnt
__device__ inline void async_cp16(const _Float16* g, _Float16* lds) {
  uint32_t l = (uint32_t)(uintptr_t)lds;
  uint64_t a = (uint64_t)(uintptr_t)g;
  asm volatile("global_load_async_to_lds_b128 %0, %1, off" :: "v"(l), "v"(a) : "memory");
}
__device__ inline void async_wait0() {
  asm volatile("s_wait_asynccnt 0" ::: "memory");
}
__device__ inline void async_wait4() {
  asm volatile("s_wait_asynccnt 4" ::: "memory");
}

// ---------------- elementwise f32 -> f16 ----------------
__global__ void f32_to_f16(const float* __restrict__ X, _Float16* __restrict__ Y, size_t n) {
  size_t i = (size_t)blockIdx.x * blockDim.x + threadIdx.x;
  if (i < n) Y[i] = (_Float16)X[i];
}

// ---------------- RMS norm (+cast to f16); input row stride may exceed L ----------------
__global__ void rmsnorm_to_f16(const float* __restrict__ X, int in_stride, int L,
                               const float* __restrict__ w, _Float16* __restrict__ Y) {
  int row = blockIdx.x;
  const float* x = X + (size_t)row * in_stride;
  __shared__ float red[256];
  float ss = 0.f;
  for (int i = threadIdx.x; i < L; i += blockDim.x) { float v = x[i]; ss += v * v; }
  red[threadIdx.x] = ss;
  __syncthreads();
  for (int s = 128; s > 0; s >>= 1) {
    if ((int)threadIdx.x < s) red[threadIdx.x] += red[threadIdx.x + s];
    __syncthreads();
  }
  float inv = rsqrtf(red[0] / (float)L + 1e-6f);
  for (int i = threadIdx.x; i < L; i += blockDim.x)
    Y[(size_t)row * L + i] = (_Float16)(x[i] * inv * w[i]);
}

// ---------------- 64x64 tiled WMMA GEMM, double-buffered async staging ----------------
__launch_bounds__(128)
__global__ void gemm_h64(const _Float16* __restrict__ A, const _Float16* __restrict__ W,
                         float* __restrict__ C, int M, int N, int K) {
  const int LDT = 40;
  __shared__ __align__(16) _Float16 sA[2][64 * 40];
  __shared__ __align__(16) _Float16 sB[2][64 * 40];
  int bm = blockIdx.y << 6, bn = blockIdx.x << 6;
  int w = threadIdx.x >> 5;
  int wm = (w >> 1) << 5, wn = (w & 1) << 5;
  v8f acc[2][2] = {};
  // 256 chunks / 128 threads => 2 iters x 2 asyncs = 4 async copies per thread per tile
  auto issue = [&](int buf, int k0) {
    for (int c = threadIdx.x; c < 256; c += 128) {
      int r = c >> 2, co = (c & 3) << 3;
      async_cp16(&A[(size_t)(bm + r) * K + k0 + co], &sA[buf][r * LDT + co]);
      async_cp16(&W[(size_t)(bn + r) * K + k0 + co], &sB[buf][r * LDT + co]);
    }
  };
  issue(0, 0);
  int it = 0;
  for (int k0 = 0; k0 < K; k0 += 32, ++it) {
    int cur = it & 1;
    bool pre = (k0 + 32 < K);
    if (pre) { issue(cur ^ 1, k0 + 32); async_wait4(); }
    else     { async_wait0(); }
    __syncthreads();
    v16h a0 = frag_a(&sA[cur][(wm) * LDT], LDT);
    v16h a1 = frag_a(&sA[cur][(wm + 16) * LDT], LDT);
    v16h b0 = frag_b(&sB[cur][(wn) * LDT], LDT);
    v16h b1 = frag_b(&sB[cur][(wn + 16) * LDT], LDT);
    acc[0][0] = __builtin_amdgcn_wmma_f32_16x16x32_f16(false, a0, false, b0, (short)0, acc[0][0], false, false);
    acc[0][1] = __builtin_amdgcn_wmma_f32_16x16x32_f16(false, a0, false, b1, (short)0, acc[0][1], false, false);
    acc[1][0] = __builtin_amdgcn_wmma_f32_16x16x32_f16(false, a1, false, b0, (short)0, acc[1][0], false, false);
    acc[1][1] = __builtin_amdgcn_wmma_f32_16x16x32_f16(false, a1, false, b1, (short)0, acc[1][1], false, false);
    __syncthreads();
  }
  int lane = threadIdx.x & 31, col = lane & 15, hi = lane >> 4;
#pragma unroll
  for (int mi = 0; mi < 2; ++mi)
#pragma unroll
    for (int ni = 0; ni < 2; ++ni)
#pragma unroll
      for (int g = 0; g < 8; ++g) {
        int row = bm + wm + mi * 16 + g + hi * 8;
        int cc  = bn + wn + ni * 16 + col;
        C[(size_t)row * N + cc] = acc[mi][ni][g];
      }
}

// ---------------- 128x128 tiled WMMA GEMM: 8 waves, 32x64/wave, double-buffered ----------------
__launch_bounds__(256)
__global__ void gemm_h128(const _Float16* __restrict__ A, const _Float16* __restrict__ W,
                          float* __restrict__ C, int M, int N, int K) {
  const int LDT = 40;
  __shared__ __align__(16) _Float16 sA[2][128 * 40];
  __shared__ __align__(16) _Float16 sB[2][128 * 40];
  int bm = blockIdx.y << 7, bn = blockIdx.x << 7;
  int w = threadIdx.x >> 5;
  int wm = (w >> 1) << 5;       // 0,32,64,96
  int wn = (w & 1) << 6;        // 0,64
  v8f acc[2][4] = {};
  // 512 chunks / 256 threads => 2 iters x 2 asyncs = 4 async copies per thread per tile
  auto issue = [&](int buf, int k0) {
    for (int c = threadIdx.x; c < 512; c += 256) {
      int r = c >> 2, co = (c & 3) << 3;
      async_cp16(&A[(size_t)(bm + r) * K + k0 + co], &sA[buf][r * LDT + co]);
      async_cp16(&W[(size_t)(bn + r) * K + k0 + co], &sB[buf][r * LDT + co]);
    }
  };
  issue(0, 0);
  int it = 0;
  for (int k0 = 0; k0 < K; k0 += 32, ++it) {
    int cur = it & 1;
    bool pre = (k0 + 32 < K);
    if (pre) { issue(cur ^ 1, k0 + 32); async_wait4(); }
    else     { async_wait0(); }
    __syncthreads();
    v16h a[2], b[4];
    a[0] = frag_a(&sA[cur][(wm) * LDT], LDT);
    a[1] = frag_a(&sA[cur][(wm + 16) * LDT], LDT);
#pragma unroll
    for (int ni = 0; ni < 4; ++ni) b[ni] = frag_b(&sB[cur][(wn + ni * 16) * LDT], LDT);
#pragma unroll
    for (int mi = 0; mi < 2; ++mi)
#pragma unroll
      for (int ni = 0; ni < 4; ++ni)
        acc[mi][ni] = __builtin_amdgcn_wmma_f32_16x16x32_f16(false, a[mi], false, b[ni],
                                                             (short)0, acc[mi][ni], false, false);
    __syncthreads();
  }
  int lane = threadIdx.x & 31, col = lane & 15, hi = lane >> 4;
#pragma unroll
  for (int mi = 0; mi < 2; ++mi)
#pragma unroll
    for (int ni = 0; ni < 4; ++ni)
#pragma unroll
      for (int g = 0; g < 8; ++g) {
        int row = bm + wm + mi * 16 + g + hi * 8;
        int cc  = bn + wn + ni * 16 + col;
        C[(size_t)row * N + cc] = acc[mi][ni][g];
      }
}

// ---------------- build q_full / k_full / v (f16) with RoPE ----------------
__global__ void assemble(const float* __restrict__ q, const float* __restrict__ kv,
                         const float* __restrict__ kvb,
                         const float* __restrict__ cosT, const float* __restrict__ sinT,
                         _Float16* __restrict__ q_full, _Float16* __restrict__ k_full,
                         _Float16* __restrict__ v_h) {
  int bt = blockIdx.x;
  int t = bt % T_;
  __shared__ float kpe_r[DR_];
  const float* kvrow = kv + (size_t)bt * (KVR_ + DR_);
  if (threadIdx.x < DR_ / 2) {
    int i = threadIdx.x;
    float x0 = kvrow[KVR_ + 2 * i], x1 = kvrow[KVR_ + 2 * i + 1];
    float c = cosT[t * (DR_ / 2) + i], s = sinT[t * (DR_ / 2) + i];
    kpe_r[2 * i]     = x0 * c - x1 * s;
    kpe_r[2 * i + 1] = x0 * s + x1 * c;
  }
  __syncthreads();
  for (int idx = threadIdx.x; idx < H_ * HD_; idx += blockDim.x) {
    int h = idx / HD_, d = idx % HD_;
    const float* qrow = q + (size_t)bt * (H_ * HD_) + (size_t)h * HD_;
    float val;
    if (d < DN_) val = qrow[d];
    else {
      int i = (d - DN_) >> 1;
      float x0 = qrow[DN_ + 2 * i], x1 = qrow[DN_ + 2 * i + 1];
      float c = cosT[t * (DR_ / 2) + i], s = sinT[t * (DR_ / 2) + i];
      val = ((d & 1) == 0) ? (x0 * c - x1 * s) : (x0 * s + x1 * c);
    }
    q_full[(size_t)bt * (H_ * HD_) + idx] = (_Float16)val;
  }
  for (int idx = threadIdx.x; idx < H_ * HD_; idx += blockDim.x) {
    int h = idx / HD_, d = idx % HD_;
    const float* kvbrow = kvb + (size_t)bt * (H_ * (DN_ + DV_)) + (size_t)h * (DN_ + DV_);
    float val = (d < DN_) ? kvbrow[d] : kpe_r[d - DN_];
    k_full[(size_t)bt * (H_ * HD_) + idx] = (_Float16)val;
  }
  for (int idx = threadIdx.x; idx < H_ * DV_; idx += blockDim.x) {
    int h = idx / DV_, d = idx % DV_;
    v_h[(size_t)bt * (H_ * DV_) + idx] =
        (_Float16)kvb[(size_t)bt * (H_ * (DN_ + DV_)) + (size_t)h * (DN_ + DV_) + DN_ + d];
  }
}

// ---------------- flash-style causal attention (per (b,h), 64 query rows/block) ----------------
__launch_bounds__(128)
__global__ void mla_attn(const _Float16* __restrict__ Qf, const _Float16* __restrict__ Kf,
                         const _Float16* __restrict__ Vh, _Float16* __restrict__ Oh) {
  const int LDQ = 200;   // 192 + 8
  const int LDV = 72;    // 64  + 8
  const int LDP = 72;
  __shared__ __align__(16) _Float16 sQ[64 * 200];
  __shared__ __align__(16) _Float16 sK[64 * 200];
  __shared__ __align__(16) _Float16 sVt[128 * 72];     // transposed: [dv][key]
  __shared__ __align__(16) _Float16 sP[4 * 16 * 72];   // per-wave P tiles
  int b = blockIdx.z, h = blockIdx.y, bx = blockIdx.x;
  int w = threadIdx.x >> 5, lane = threadIdx.x & 31;
  int col = lane & 15, hi = lane >> 4;
  size_t tok0 = (size_t)b * T_;

  // stage Q tile (64 x 192) via async LDS copies
  for (int c = threadIdx.x; c < 1536; c += 128) {
    int r = c / 24, co = (c % 24) << 3;
    async_cp16(&Qf[(tok0 + (size_t)bx * 64 + r) * (H_ * HD_) + (size_t)h * HD_ + co],
               &sQ[r * LDQ + co]);
  }
  async_wait0();
  __syncthreads();
  // Q fragments held in registers for the whole key loop (6 x v16h = 48 VGPRs)
  v16h qa[6];
#pragma unroll
  for (int kk = 0; kk < 6; ++kk) qa[kk] = frag_a(&sQ[(w * 16) * LDQ + kk * 32], LDQ);

  float m_i[8], l_i[8];
  v8f o[8] = {};
#pragma unroll
  for (int g = 0; g < 8; ++g) { m_i[g] = -1e30f; l_i[g] = 0.f; }
  const float scale = 0.0721687836487032f;  // (DN+DR)^-0.5

  for (int kt = 0; kt <= bx; ++kt) {
    __syncthreads();
    // stage K tile (64 x 192) asynchronously; prefetch next tile into L2
    for (int c = threadIdx.x; c < 1536; c += 128) {
      int r = c / 24, co = (c % 24) << 3;
      const _Float16* gk = &Kf[(tok0 + (size_t)kt * 64 + r) * (H_ * HD_) + (size_t)h * HD_ + co];
      async_cp16(gk, &sK[r * LDQ + co]);
      if (kt < bx) __builtin_prefetch(gk + (size_t)64 * (H_ * HD_), 0, 0);
    }
    // stage V tile transposed (dv-major) via registers
    for (int c = threadIdx.x; c < 1024; c += 128) {
      int key = c >> 4, d0 = (c & 15) << 3;
      const _Float16* gv = &Vh[(tok0 + (size_t)kt * 64 + key) * (H_ * DV_) + (size_t)h * DV_ + d0];
      v8h v = *(const v8h*)gv;
      if (kt < bx) __builtin_prefetch(gv + (size_t)64 * (H_ * DV_), 0, 0);
#pragma unroll
      for (int i = 0; i < 8; ++i) sVt[(d0 + i) * LDV + key] = v[i];
    }
    async_wait0();
    __syncthreads();

    // S (16x64 per wave) = Q_tile @ K_tile^T
    v8f s[4] = {};
#pragma unroll
    for (int kk = 0; kk < 6; ++kk) {
#pragma unroll
      for (int ni = 0; ni < 4; ++ni) {
        v16h bf = frag_b(&sK[(ni * 16) * LDQ + kk * 32], LDQ);
        s[ni] = __builtin_amdgcn_wmma_f32_16x16x32_f16(false, qa[kk], false, bf, (short)0, s[ni], false, false);
      }
    }
    // scale + causal mask (only diagonal key tile can clip)
#pragma unroll
    for (int ni = 0; ni < 4; ++ni)
#pragma unroll
      for (int g = 0; g < 8; ++g) {
        float v = s[ni][g] * scale;
        if (kt == bx) {
          int rg = bx * 64 + w * 16 + g + hi * 8;
          int cg = kt * 64 + ni * 16 + col;
          if (cg > rg) v = -1e30f;
        }
        s[ni][g] = v;
      }
    // online softmax; rows live in 16-lane halves, reduce with xor 1/2/4/8
    float alpha[8];
#pragma unroll
    for (int g = 0; g < 8; ++g) {
      float tm = fmaxf(fmaxf(s[0][g], s[1][g]), fmaxf(s[2][g], s[3][g]));
      tm = fmaxf(tm, __shfl_xor(tm, 1));
      tm = fmaxf(tm, __shfl_xor(tm, 2));
      tm = fmaxf(tm, __shfl_xor(tm, 4));
      tm = fmaxf(tm, __shfl_xor(tm, 8));
      float mn = fmaxf(m_i[g], tm);
      alpha[g] = __expf(m_i[g] - mn);
      m_i[g] = mn;
      float rs = 0.f;
#pragma unroll
      for (int ni = 0; ni < 4; ++ni) {
        float p = __expf(s[ni][g] - mn);
        s[ni][g] = p;
        rs += p;
      }
      rs += __shfl_xor(rs, 1);
      rs += __shfl_xor(rs, 2);
      rs += __shfl_xor(rs, 4);
      rs += __shfl_xor(rs, 8);
      l_i[g] = l_i[g] * alpha[g] + rs;
    }
    // rescale running O
#pragma unroll
    for (int ni = 0; ni < 8; ++ni)
#pragma unroll
      for (int g = 0; g < 8; ++g) o[ni][g] *= alpha[g];
    // P (C layout) -> per-wave LDS -> A fragments
    _Float16* pP = &sP[w * 16 * LDP];
#pragma unroll
    for (int ni = 0; ni < 4; ++ni)
#pragma unroll
      for (int g = 0; g < 8; ++g)
        pP[(g + hi * 8) * LDP + ni * 16 + col] = (_Float16)s[ni][g];
    asm volatile("s_wait_dscnt 0" ::: "memory");   // wave-local LDS RAW fence
    // O += P @ V
#pragma unroll
    for (int kk = 0; kk < 2; ++kk) {
      v16h a = frag_a(&pP[kk * 32], LDP);
#pragma unroll
      for (int ni = 0; ni < 8; ++ni) {
        v16h bf = frag_b(&sVt[(ni * 16) * LDV + kk * 32], LDV);
        o[ni] = __builtin_amdgcn_wmma_f32_16x16x32_f16(false, a, false, bf, (short)0, o[ni], false, false);
      }
    }
  }
  // epilogue: O /= l, write f16 (B,T,H*DV)
#pragma unroll
  for (int g = 0; g < 8; ++g) l_i[g] = 1.f / l_i[g];
  size_t obase = (tok0 + (size_t)bx * 64 + w * 16) * (size_t)(H_ * DV_) + (size_t)h * DV_;
#pragma unroll
  for (int ni = 0; ni < 8; ++ni)
#pragma unroll
    for (int g = 0; g < 8; ++g) {
      int row = g + hi * 8;
      Oh[obase + (size_t)row * (H_ * DV_) + ni * 16 + col] = (_Float16)(o[ni][g] * l_i[g]);
    }
}

// ---------------- launch ----------------
extern "C" void kernel_launch(void* const* d_in, const int* in_sizes, int n_in,
                              void* d_out, int out_size, void* d_ws, size_t ws_size,
                              hipStream_t stream) {
  const float* x       = (const float*)d_in[0];
  const float* wq_a    = (const float*)d_in[1];
  const float* q_normw = (const float*)d_in[2];
  const float* wq_b    = (const float*)d_in[3];
  const float* wkv_a   = (const float*)d_in[4];
  const float* kv_normw= (const float*)d_in[5];
  const float* wkv_b   = (const float*)d_in[6];
  const float* wo      = (const float*)d_in[7];
  const float* cosT    = (const float*)d_in[8];
  const float* sinT    = (const float*)d_in[9];

  char* wp = (char*)d_ws;
  auto alloc = [&](size_t bytes) -> void* {
    void* p = (void*)wp;
    wp += (bytes + 255) & ~(size_t)255;
    return p;
  };
  _Float16* xh        = (_Float16*)alloc((size_t)BT_ * DIM_ * 2);
  _Float16* wq_a_h    = (_Float16*)alloc((size_t)QR_ * DIM_ * 2);
  _Float16* wq_b_h    = (_Float16*)alloc((size_t)H_ * HD_ * QR_ * 2);
  _Float16* wkv_a_h   = (_Float16*)alloc((size_t)(KVR_ + DR_) * DIM_ * 2);
  _Float16* wkv_b_h   = (_Float16*)alloc((size_t)H_ * (DN_ + DV_) * KVR_ * 2);
  _Float16* wo_h      = (_Float16*)alloc((size_t)DIM_ * H_ * DV_ * 2);
  float*    q_lat     = (float*)   alloc((size_t)BT_ * QR_ * 4);
  float*    kvbuf     = (float*)   alloc((size_t)BT_ * (KVR_ + DR_) * 4);
  _Float16* q_lat_h   = (_Float16*)alloc((size_t)BT_ * QR_ * 2);
  _Float16* kv_norm_h = (_Float16*)alloc((size_t)BT_ * KVR_ * 2);
  float*    qbuf      = (float*)   alloc((size_t)BT_ * H_ * HD_ * 4);
  float*    kvb       = (float*)   alloc((size_t)BT_ * H_ * (DN_ + DV_) * 4);
  _Float16* q_full    = (_Float16*)alloc((size_t)BT_ * H_ * HD_ * 2);
  _Float16* k_full    = (_Float16*)alloc((size_t)BT_ * H_ * HD_ * 2);
  _Float16* v_h       = (_Float16*)alloc((size_t)BT_ * H_ * DV_ * 2);
  _Float16* attn_h    = (_Float16*)alloc((size_t)BT_ * H_ * DV_ * 2);

  auto cvt = [&](const float* src, _Float16* dst, size_t n) {
    f32_to_f16<<<dim3((unsigned)((n + 255) / 256)), dim3(256), 0, stream>>>(src, dst, n);
  };
  auto gemm = [&](const _Float16* A, const _Float16* W, float* C, int M, int N, int K) {
    if ((N & 127) == 0)
      gemm_h128<<<dim3(N / 128, M / 128), dim3(256), 0, stream>>>(A, W, C, M, N, K);
    else
      gemm_h64<<<dim3(N / 64, M / 64), dim3(128), 0, stream>>>(A, W, C, M, N, K);
  };

  cvt(x,     xh,      (size_t)BT_ * DIM_);
  cvt(wq_a,  wq_a_h,  (size_t)QR_ * DIM_);
  cvt(wq_b,  wq_b_h,  (size_t)H_ * HD_ * QR_);
  cvt(wkv_a, wkv_a_h, (size_t)(KVR_ + DR_) * DIM_);
  cvt(wkv_b, wkv_b_h, (size_t)H_ * (DN_ + DV_) * KVR_);
  cvt(wo,    wo_h,    (size_t)DIM_ * H_ * DV_);

  gemm(xh, wq_a_h,  q_lat, BT_, QR_,         DIM_);   // x @ wq_a^T
  gemm(xh, wkv_a_h, kvbuf, BT_, KVR_ + DR_,  DIM_);   // x @ wkv_a^T  (N=576 -> 64-tile path)

  rmsnorm_to_f16<<<dim3(BT_), dim3(256), 0, stream>>>(q_lat, QR_,        QR_,  q_normw,  q_lat_h);
  rmsnorm_to_f16<<<dim3(BT_), dim3(256), 0, stream>>>(kvbuf, KVR_ + DR_, KVR_, kv_normw, kv_norm_h);

  gemm(q_lat_h,   wq_b_h,  qbuf, BT_, H_ * HD_,          QR_);   // -> q
  gemm(kv_norm_h, wkv_b_h, kvb,  BT_, H_ * (DN_ + DV_),  KVR_);  // -> kvb

  assemble<<<dim3(BT_), dim3(256), 0, stream>>>(qbuf, kvbuf, kvb, cosT, sinT,
                                                q_full, k_full, v_h);

  mla_attn<<<dim3(T_ / 64, H_, B_), dim3(128), 0, stream>>>(q_full, k_full, v_h, attn_h);

  gemm(attn_h, wo_h, (float*)d_out, BT_, DIM_, H_ * DV_);        // attn @ wo^T
}
